// GCN_regression_Net_63376537420022
// MI455X (gfx1250) — matrix operand order repaired and checked by reading, not compiled, not run on hardware
//
#include <hip/hip_runtime.h>

// ---------- types ----------
typedef __attribute__((ext_vector_type(16))) __bf16          v16bf;
typedef __attribute__((ext_vector_type(8)))  float           v8f;
typedef __attribute__((ext_vector_type(4)))  unsigned int    u32x4;
typedef __attribute__((ext_vector_type(4)))  unsigned short  u16x4;
typedef __attribute__((ext_vector_type(4)))  float           f32x4;

union ABPack { u32x4 u[2]; v16bf v; };

__device__ __forceinline__ unsigned short f2bf(float f) {
  unsigned u = __builtin_bit_cast(unsigned, f);
  u = (u + 0x7FFFu + ((u >> 16) & 1u)) >> 16;   // round-to-nearest-even
  return (unsigned short)u;
}
__device__ __forceinline__ float bf2f(unsigned short h) {
  unsigned u = ((unsigned)h) << 16;
  return __builtin_bit_cast(float, u);
}

// ---------- fp32 -> bf16 row-major activation conversion ----------
__global__ __launch_bounds__(256) void k_cvt_bf16(const float* __restrict__ x,
                                                  unsigned short* __restrict__ xb, long n4) {
  long i = (long)blockIdx.x * 256 + threadIdx.x;
  if (i >= n4) return;
  f32x4 v = *(const f32x4*)(x + i * 4);
  u16x4 r = { f2bf(v.x), f2bf(v.y), f2bf(v.z), f2bf(v.w) };
  *(u16x4*)(xb + i * 4) = r;
}

// ---------- fill ----------
__global__ __launch_bounds__(256) void k_fill(float* __restrict__ p, float val, int n) {
  int i = blockIdx.x * 256 + threadIdx.x;
  if (i < n) p[i] = val;
}

// ---------- degree accumulation over edges (self-loop "1" preset by k_fill) ----------
__global__ __launch_bounds__(256) void k_deg(const int* __restrict__ dst, const float* __restrict__ ew,
                                             float* __restrict__ deg, int ne) {
  int e = blockIdx.x * 256 + threadIdx.x;
  if (e < ne)
    __hip_atomic_fetch_add(&deg[dst[e]], ew[e], __ATOMIC_RELAXED, __HIP_MEMORY_SCOPE_AGENT);
}

// ---------- deg -> dinv (in place) ----------
__global__ __launch_bounds__(256) void k_rsqrt(float* __restrict__ p, int n) {
  int i = blockIdx.x * 256 + threadIdx.x;
  if (i < n) { float d = p[i]; p[i] = d > 0.f ? rsqrtf(d) : 0.f; }
}

// ---------- pack W[128][128] fp32 -> per-lane WMMA B layout, bf16 ----------
// wp[((nt*4 + kk)*32 + lane)*16 + e] = bf16( W[(kk*32 + (lane>>4)*16 + e)*128 + nt*16 + (lane&15)] )
__global__ __launch_bounds__(256) void k_packw(const float* __restrict__ W,
                                               unsigned short* __restrict__ wp) {
  int idx = blockIdx.x * 256 + threadIdx.x;         // exactly 16384 threads launched
  int e    = idx & 15;
  int lane = (idx >> 4) & 31;
  int kk   = (idx >> 9) & 3;
  int nt   = idx >> 11;                             // 0..7
  int k = kk * 32 + (lane >> 4) * 16 + e;
  int n = nt * 16 + (lane & 15);
  wp[idx] = f2bf(W[k * 128 + n]);
}

// ---------- GEMM + fused self-loop scale ----------
// xw = xb @ W (bf16 in, f32 accumulate);  acc = dinv[row]^2 * xw  (self-loop term)
// block = 256 thr = 8 waves; wave nt owns N-tile nt, B held in registers for all K.
// A is double-buffered across M-tiles: loads for tile t+1 are issued before the
// WMMAs of tile t, so each WMMA group waits only on loads issued a full tile ago.
#define MT_PER_BLK 8
__global__ __launch_bounds__(256) void k_gemm(const unsigned short* __restrict__ xb,
                                              const unsigned short* __restrict__ wp,
                                              const float* __restrict__ dinv,
                                              float* __restrict__ xw,
                                              float* __restrict__ acc,
                                              int nrows, int mtiles) {
  const int lane = threadIdx.x & 31;
  const int nt   = threadIdx.x >> 5;
  const int m    = lane & 15;
  const int hi   = lane >> 4;
  const int col  = nt * 16 + m;

  // hoist B: 4 K-chunks x 32 bytes/lane, loop-invariant
  ABPack b[4];
  const unsigned short* bp0 = wp + ((nt * 4) * 32 + lane) * 16;
#pragma unroll
  for (int kk = 0; kk < 4; ++kk) {
    b[kk].u[0] = *(const u32x4*)(bp0 + kk * 512);
    b[kk].u[1] = *(const u32x4*)(bp0 + kk * 512 + 8);
  }

  const int mt0 = blockIdx.x * MT_PER_BLK;
  int ntiles = mtiles - mt0;
  if (ntiles <= 0) return;
  if (ntiles > MT_PER_BLK) ntiles = MT_PER_BLK;

  auto loadA = [&](ABPack* A, int t) {
    const int row0 = (mt0 + t) * 16;
    int arow = row0 + m; if (arow >= nrows) arow = nrows - 1;   // EXEC all-1 for WMMA
    const unsigned short* ap0 = xb + (long)arow * 128 + hi * 16;
#pragma unroll
    for (int kk = 0; kk < 4; ++kk) {
      A[kk].u[0] = *(const u32x4*)(ap0 + kk * 32);       // K = kk*32 + hi*16     .. +7
      A[kk].u[1] = *(const u32x4*)(ap0 + kk * 32 + 8);   // K = kk*32 + hi*16 + 8 .. +7
    }
  };

  auto computeStore = [&](const ABPack* A, int t) {
    v8f c = {};
#pragma unroll
    for (int kk = 0; kk < 4; ++kk)
      c = __builtin_amdgcn_wmma_f32_16x16x32_bf16(false, A[kk].v, false, b[kk].v,
                                                  (short)0, c, false, false);
    const int row0 = (mt0 + t) * 16;
    if (row0 + 16 <= nrows) {                // fast path: full tile, branch-free stores
      f32x4 d0 = *(const f32x4*)(dinv + row0 + hi * 8);
      f32x4 d1 = *(const f32x4*)(dinv + row0 + hi * 8 + 4);
#pragma unroll
      for (int v = 0; v < 8; ++v) {
        const int row = row0 + v + hi * 8;   // C/D layout: VGPR v -> M = v + 8*hi
        const float dv = (v < 4) ? d0[v] : d1[v - 4];
        const float val = c[v];
        xw [(long)row * 128 + col] = val;
        acc[(long)row * 128 + col] = val * dv * dv;
      }
    } else {                                 // rare partial tail tile
#pragma unroll
      for (int v = 0; v < 8; ++v) {
        const int row = row0 + v + hi * 8;
        if (row < nrows) {
          const float dv = dinv[row];
          const float val = c[v];
          xw [(long)row * 128 + col] = val;
          acc[(long)row * 128 + col] = val * dv * dv;
        }
      }
    }
  };

  ABPack A0[4], A1[4];
  loadA(A0, 0);
  for (int t = 0; t < ntiles; t += 2) {
    if (t + 1 < ntiles) loadA(A1, t + 1);    // prefetch next tile before consuming A0
    computeStore(A0, t);
    if (t + 1 < ntiles) {
      if (t + 2 < ntiles) loadA(A0, t + 2);  // prefetch before consuming A1
      computeStore(A1, t + 1);
    }
  }
}

// ---------- edge aggregation: one wave per edge, lane covers 4 features ----------
__global__ __launch_bounds__(256) void k_agg(const int* __restrict__ src, const int* __restrict__ dst,
                                             const float* __restrict__ ew, const float* __restrict__ dinv,
                                             const float* __restrict__ xw, float* __restrict__ acc, int ne) {
  const int lane = threadIdx.x & 31;
  const long e = (long)blockIdx.x * 8 + (threadIdx.x >> 5);
  if (e >= ne) return;
  const int r = src[e], cn = dst[e];
  const float norm = dinv[r] * ew[e] * dinv[cn];
  f32x4 v = *(const f32x4*)(xw + (long)r * 128 + lane * 4);
  float* ap = acc + (long)cn * 128 + lane * 4;
  __hip_atomic_fetch_add(ap + 0, norm * v.x, __ATOMIC_RELAXED, __HIP_MEMORY_SCOPE_AGENT);
  __hip_atomic_fetch_add(ap + 1, norm * v.y, __ATOMIC_RELAXED, __HIP_MEMORY_SCOPE_AGENT);
  __hip_atomic_fetch_add(ap + 2, norm * v.z, __ATOMIC_RELAXED, __HIP_MEMORY_SCOPE_AGENT);
  __hip_atomic_fetch_add(ap + 3, norm * v.w, __ATOMIC_RELAXED, __HIP_MEMORY_SCOPE_AGENT);
}

// ---------- bias + ReLU + bf16 for next layer's WMMA ----------
__global__ __launch_bounds__(256) void k_final(const float* __restrict__ acc,
                                               const float* __restrict__ bias,
                                               unsigned short* __restrict__ xb, long n4) {
  long i = (long)blockIdx.x * 256 + threadIdx.x;
  if (i >= n4) return;
  int c = ((int)i & 31) * 4;
  f32x4 v = *(const f32x4*)(acc + i * 4);
  u16x4 r = { f2bf(fmaxf(v.x + bias[c + 0], 0.f)),
              f2bf(fmaxf(v.y + bias[c + 1], 0.f)),
              f2bf(fmaxf(v.z + bias[c + 2], 0.f)),
              f2bf(fmaxf(v.w + bias[c + 3], 0.f)) };
  *(u16x4*)(xb + i * 4) = r;
}

// ---------- out init with lin_b ----------
__global__ __launch_bounds__(256) void k_outinit(float* __restrict__ out,
                                                 const float* __restrict__ linb, int n) {
  int i = blockIdx.x * 256 + threadIdx.x;
  if (i < n) out[i] = linb[0];
}

// ---------- fused pool + linear: out[batch[i]] += dot(x3[i], lin_w) ----------
__global__ __launch_bounds__(256) void k_pool(const unsigned short* __restrict__ xb,
                                              const int* __restrict__ batch,
                                              const float* __restrict__ linw,
                                              float* __restrict__ out, int nnodes) {
  const int lane = threadIdx.x & 31;
  const long node = (long)blockIdx.x * 8 + (threadIdx.x >> 5);
  if (node >= nnodes) return;
  u16x4 h = *(const u16x4*)(xb + node * 128 + lane * 4);
  f32x4 w = *(const f32x4*)(linw + lane * 4);
  float s = bf2f(h.x) * w.x + bf2f(h.y) * w.y + bf2f(h.z) * w.z + bf2f(h.w) * w.w;
#pragma unroll
  for (int off = 16; off > 0; off >>= 1) s += __shfl_down(s, off, 32);
  if (lane == 0)
    __hip_atomic_fetch_add(&out[batch[node]], s, __ATOMIC_RELAXED, __HIP_MEMORY_SCOPE_AGENT);
}

extern "C" void kernel_launch(void* const* d_in, const int* in_sizes, int n_in,
                              void* d_out, int out_size, void* d_ws, size_t ws_size,
                              hipStream_t stream) {
  const float* X     = (const float*)d_in[0];
  const int*   ei    = (const int*)d_in[1];
  const float* ew    = (const float*)d_in[2];
  const int*   batch = (const int*)d_in[3];
  const float* Ws[3] = { (const float*)d_in[4], (const float*)d_in[6], (const float*)d_in[8] };
  const float* bs[3] = { (const float*)d_in[5], (const float*)d_in[7], (const float*)d_in[9] };
  const float* linw  = (const float*)d_in[10];
  const float* linb  = (const float*)d_in[11];
  float* out = (float*)d_out;

  const int nnodes = in_sizes[0] / 128;
  const int nedges = in_sizes[2];
  const int* src = ei;            // edge_index[0] = sources
  const int* dst = ei + nedges;   // edge_index[1] = targets

  // workspace layout
  char* w = (char*)d_ws;
  unsigned short* xb   = (unsigned short*)w;                                   // bf16 activations
  float*          xw   = (float*)(w + (size_t)nnodes * 256);                   // x @ W (fp32)
  float*          acc  = (float*)(w + (size_t)nnodes * 768);                   // aggregation acc
  float*          dinv = (float*)(w + (size_t)nnodes * 1280);                  // deg -> rsqrt
  unsigned short* wp   = (unsigned short*)(w + (size_t)nnodes * 1284);         // packed W (32 KB)

  const long n4     = (long)nnodes * 32;          // groups of 4 features
  const int  nblk4  = (int)((n4 + 255) / 256);
  const int  mtiles = (nnodes + 15) / 16;
  const int  gblk   = (mtiles + MT_PER_BLK - 1) / MT_PER_BLK;
  const int  eblk   = (nedges + 255) / 256;
  const int  ewave  = (nedges + 7) / 8;
  const int  nblk   = (nnodes + 255) / 256;

  // X -> bf16 once; normalization coefficients once
  k_cvt_bf16<<<nblk4, 256, 0, stream>>>(X, xb, n4);
  k_fill    <<<nblk, 256, 0, stream>>>(dinv, 1.0f, nnodes);           // self-loop weight
  k_deg     <<<eblk, 256, 0, stream>>>(dst, ew, dinv, nedges);
  k_rsqrt   <<<nblk, 256, 0, stream>>>(dinv, nnodes);

  for (int l = 0; l < 3; ++l) {
    k_packw<<<64, 256, 0, stream>>>(Ws[l], wp);
    k_gemm <<<gblk, 256, 0, stream>>>(xb, wp, dinv, xw, acc, nnodes, mtiles);
    k_agg  <<<ewave, 256, 0, stream>>>(src, dst, ew, dinv, xw, acc, nedges);
    k_final<<<nblk4, 256, 0, stream>>>(acc, bs[l], xb, n4);
  }

  k_outinit<<<(out_size + 255) / 256, 256, 0, stream>>>(out, linb, out_size);
  k_pool   <<<(nnodes + 7) / 8, 256, 0, stream>>>(xb, batch, linw, out, nnodes);
}
